// DCFNTM_11536282157643
// MI455X (gfx1250) — matrix operand order repaired and checked by reading, not compiled, not run on hardware
//
#include <hip/hip_runtime.h>
#include <hip/hip_bf16.h>

// ---------------- problem constants ----------------
#define Bb 8
#define Tt 16
#define BT 128
#define IMG 128
#define CF 32
#define WO 32
#define HO 32
#define NSLOT 1024
#define DH 256
#define WF 17          // WO/2 + 1
#define LD 33          // padded FFT row stride

typedef __attribute__((ext_vector_type(16))) _Float16 v16h;
typedef __attribute__((ext_vector_type(8)))  _Float16 v8h;
typedef __attribute__((ext_vector_type(8)))  float    v8f;

// low 32 bits of an LDS-aperture flat address are the LDS byte offset
__device__ __forceinline__ unsigned lds_addr32(const void* p) {
  return (unsigned)(size_t)p;
}

// ---------------- WMMA tile loaders (CDNA5 16x16x32 f16 layouts) ----------------
// A 16x32 f16, row-major ld=32. Lane fragment K sets are two runs of 8:
// lanes 0-15: K[0..7] then K[16..23]; lanes 16-31: K[8..15] then K[24..31]
__device__ __forceinline__ v16h wmma_load_a(const _Float16* sm) {
  const int lane = threadIdx.x & 31;
  const int r = lane & 15, hg = lane >> 4;
  const v8h* row = (const v8h*)(sm + r * 32);     // 4 x v8h per row
  v8h lo = row[hg];                               // K hg*8 .. hg*8+7
  v8h hi = row[2 + hg];                           // K 16+hg*8 ..
  return __builtin_shufflevector(lo, hi, 0, 1, 2, 3, 4, 5, 6, 7,
                                 8, 9, 10, 11, 12, 13, 14, 15);
}
// B 32x16: stored N-major in LDS as [n][k] (ld=32). Lane col n = (lane&15)+col0,
// K = hg*8 .. hg*8+7 then 16+hg*8 .. 16+hg*8+7 (same as A).
__device__ __forceinline__ v16h wmma_load_b(const _Float16* sm, int col0) {
  const int lane = threadIdx.x & 31;
  const int n = (lane & 15) + col0, hg = lane >> 4;
  const v8h* row = (const v8h*)(sm + n * 32);
  v8h lo = row[hg];
  v8h hi = row[2 + hg];
  return __builtin_shufflevector(lo, hi, 0, 1, 2, 3, 4, 5, 6, 7,
                                 8, 9, 10, 11, 12, 13, 14, 15);
}

// ---------------- conv1: 3->32, stride2, SAME, 128->64; out NHWC f16 ----------------
__global__ void conv1_kernel(const float* __restrict__ x_i, const float* __restrict__ z_i,
                             const float* __restrict__ w, const float* __restrict__ bias,
                             _Float16* __restrict__ y1x, _Float16* __restrict__ y1z) {
  __shared__ alignas(16) _Float16 Asm[16 * 32];
  __shared__ alignas(16) _Float16 Bsm[32 * 32];   // [o][k]
  const int tid  = threadIdx.x;            // 32 threads = 1 wave
  const int img  = blockIdx.y;             // 0..127
  const int tile = blockIdx.x;             // 0..255 (16 px each, row-major 64x64)
  const float* src = (blockIdx.z == 0 ? x_i : z_i) + (size_t)img * 3 * IMG * IMG;
  _Float16* dst = (blockIdx.z == 0 ? y1x : y1z) + (size_t)img * 64 * 64 * 32;

  // weights: Bsm[o][k], k = ci*9 + kh*3 + kw (27 real, pad 32)
  {
    const int o = tid;
#pragma unroll
    for (int k = 0; k < 32; ++k)
      Bsm[o * 32 + k] = (_Float16)(k < 27 ? w[o * 27 + k] : 0.0f);
  }
  const int pix0 = tile * 16;
  const int oy   = pix0 >> 6;
  const int ox0  = pix0 & 63;
  __builtin_prefetch(src + (size_t)(2 * oy) * IMG, 0, 1);
  // im2col A tile: Asm[p][k]  (SAME pad: pad_lo=0, pad_hi=1)
#pragma unroll
  for (int i = 0; i < 16; ++i) {
    int idx = tid * 16 + i;
    int p = idx >> 5, k = idx & 31;
    float v = 0.0f;
    if (k < 27) {
      int ci = k / 9, rem = k % 9, kh = rem / 3, kw = rem % 3;
      int iy = 2 * oy + kh;
      int ix = 2 * (ox0 + p) + kw;
      if (iy < IMG && ix < IMG) v = src[ci * (IMG * IMG) + iy * IMG + ix];
    }
    Asm[p * 32 + k] = (_Float16)v;
  }
  __syncthreads();
  v16h a  = wmma_load_a(Asm);
  v16h b0 = wmma_load_b(Bsm, 0);
  v16h b1 = wmma_load_b(Bsm, 16);
  v8f c0 = {}; v8f c1 = {};
  c0 = __builtin_amdgcn_wmma_f32_16x16x32_f16(false, a, false, b0, (short)0, c0, false, false);
  c1 = __builtin_amdgcn_wmma_f32_16x16x32_f16(false, a, false, b1, (short)0, c1, false, false);
  const int lane = tid & 31, col = lane & 15, hg = lane >> 4;
#pragma unroll
  for (int i = 0; i < 8; ++i) {
    int p = i + hg * 8;
    float v0 = c0[i] + bias[col];      v0 = v0 > 0.0f ? v0 : 0.0f;
    float v1 = c1[i] + bias[col + 16]; v1 = v1 > 0.0f ? v1 : 0.0f;
    dst[(size_t)(pix0 + p) * 32 + col]      = (_Float16)v0;
    dst[(size_t)(pix0 + p) * 32 + col + 16] = (_Float16)v1;
  }
}

// ---------------- conv2: 32->32, stride2, SAME, 64->32; in NHWC f16, out NHWC f32 ----------------
__global__ void conv2_kernel(const _Float16* __restrict__ y1x, const _Float16* __restrict__ y1z,
                             const float* __restrict__ w, const float* __restrict__ bias,
                             float* __restrict__ xf2, float* __restrict__ zf2) {
  __shared__ alignas(16) _Float16 Asm[16 * 32];
  __shared__ alignas(16) _Float16 Bsm[9][32 * 32];   // [khkw][o][c]
  const int tid  = threadIdx.x;
  const int img  = blockIdx.y;
  const int tile = blockIdx.x;           // 0..63 (16 px each, row-major 32x32)
  const _Float16* src = (blockIdx.z == 0 ? y1x : y1z) + (size_t)img * 64 * 64 * 32;
  float* dst = (blockIdx.z == 0 ? xf2 : zf2) + (size_t)img * NSLOT * 32;

  // weights: Bsm[s][o][c], s = kh*3+kw; fragment reads row o, K = c
  for (int idx = tid; idx < 9 * 32 * 32; idx += 32) {
    int s = idx >> 10, rem = idx & 1023, o = rem >> 5, c = rem & 31;
    int kh = s / 3, kw = s % 3;
    Bsm[s][o * 32 + c] = (_Float16)w[((o * 32 + c) * 3 + kh) * 3 + kw];
  }
  const int pix0 = tile * 16;
  const int oy   = pix0 >> 5;
  const int ox0  = pix0 & 31;
  __builtin_prefetch(src + (size_t)(2 * oy) * 64 * 32, 0, 1);
  v8f c0 = {}, c1 = {};
  const int p    = tid >> 1;             // 2 threads per pixel
  const int phal = tid & 1;
  __syncthreads();
  for (int s = 0; s < 9; ++s) {
    const int kh = s / 3, kw = s % 3;
    // A fill: vectorized NHWC copy (32 contiguous halfs per pixel)
    {
      int iy = 2 * oy + kh;
      int ix = 2 * (ox0 + p) + kw;
      v8h* ap = (v8h*)(Asm + p * 32);
      if (iy < 64 && ix < 64) {
        const v8h* gp = (const v8h*)(src + ((size_t)(iy * 64 + ix)) * 32);
        ap[phal * 2]     = gp[phal * 2];
        ap[phal * 2 + 1] = gp[phal * 2 + 1];
      } else {
        v8h z = {};
        ap[phal * 2]     = z;
        ap[phal * 2 + 1] = z;
      }
    }
    __syncthreads();
    v16h a  = wmma_load_a(Asm);
    v16h b0 = wmma_load_b(&Bsm[s][0], 0);
    v16h b1 = wmma_load_b(&Bsm[s][0], 16);
    c0 = __builtin_amdgcn_wmma_f32_16x16x32_f16(false, a, false, b0, (short)0, c0, false, false);
    c1 = __builtin_amdgcn_wmma_f32_16x16x32_f16(false, a, false, b1, (short)0, c1, false, false);
    __syncthreads();
  }
  const int lane = tid & 31, col = lane & 15, hg = lane >> 4;
#pragma unroll
  for (int i = 0; i < 8; ++i) {
    int pp = i + hg * 8;
    float v0 = c0[i] + bias[col];      v0 = v0 > 0.0f ? v0 : 0.0f;
    float v1 = c1[i] + bias[col + 16]; v1 = v1 > 0.0f ? v1 : 0.0f;
    dst[(size_t)(pix0 + pp) * 32 + col]      = v0;
    dst[(size_t)(pix0 + pp) * 32 + col + 16] = v1;
  }
}

// ---------------- per-(b,t) spatial mean of x features ----------------
__global__ void xmean_kernel(const float* __restrict__ xf2, float* __restrict__ xmean) {
  __shared__ float part[256];
  const int s = blockIdx.x, tid = threadIdx.x;
  const int m = tid & 31, g = tid >> 5;
  const float* base = xf2 + (size_t)s * NSLOT * 32;
  float acc = 0.0f;
  for (int n = g; n < NSLOT; n += 8) acc += base[n * 32 + m];
  part[tid] = acc;
  __syncthreads();
  if (g == 0) {
    float t = 0.0f;
    for (int gg = 0; gg < 8; ++gg) t += part[gg * 32 + m];
    xmean[s * 32 + m] = t * (1.0f / (float)NSLOT);
  }
}

// ---------------- block reductions ----------------
__device__ __forceinline__ float block_reduce_sum(float v, float* red, int tid) {
  red[tid] = v; __syncthreads();
  for (int s = 128; s > 0; s >>= 1) { if (tid < s) red[tid] += red[tid + s]; __syncthreads(); }
  float r = red[0]; __syncthreads();
  return r;
}
__device__ __forceinline__ float block_reduce_max(float v, float* red, int tid) {
  red[tid] = v; __syncthreads();
  for (int s = 128; s > 0; s >>= 1) { if (tid < s) red[tid] = fmaxf(red[tid], red[tid + s]); __syncthreads(); }
  float r = red[0]; __syncthreads();
  return r;
}

// ---------------- NTM scan: one block per batch, memory resident in LDS ----------------
#define SCAN_SMEM_FLOATS (32768 + 1024 + 256 + 32 * 4 + 256 + 256 + 8)
__global__ void scan_kernel(const float* __restrict__ xf2,
                            const float* __restrict__ Wk, const float* __restrict__ bk,
                            const float* __restrict__ Wbeta, const float* __restrict__ bbeta,
                            const float* __restrict__ Wh, const float* __restrict__ bh,
                            const float* __restrict__ We, const float* __restrict__ Wa,
                            const float* __restrict__ Wg,
                            const float* __restrict__ xmean, float* __restrict__ c_seq) {
  extern __shared__ float smem[];
  float* C    = smem;                 // [1024][32]
  float* W    = C + 32768;            // [1024]
  float* H    = W + 1024;             // [256]
  float* K    = H + 256;              // [32]
  float* R    = K + 32;               // [32]
  float* E    = R + 32;               // [32]
  float* AA   = E + 32;               // [32]
  float* PART = AA + 32;              // [8][32]
  float* RED  = PART + 256;           // [256]
  float* SC   = RED + 256;            // [8]

  const int b = blockIdx.x, tid = threadIdx.x;
  const float* xb = xf2 + (size_t)b * Tt * NSLOT * 32;
  float* cb = c_seq + (size_t)b * Tt * NSLOT * 32;

  H[tid] = 0.0f;
  // c0 = layernorm(xf[:,0]) over all N*M elements
  float s1 = 0.0f, s2 = 0.0f;
  for (int i = tid; i < 32768; i += 256) { float v = xb[i]; s1 += v; s2 += v * v; }
  float tot  = block_reduce_sum(s1, RED, tid);
  float tot2 = block_reduce_sum(s2, RED, tid);
  float mu  = tot * (1.0f / 32768.0f);
  float var = tot2 * (1.0f / 32768.0f) - mu * mu;
  float ninv = rsqrtf(var + 1e-5f);
  for (int i = tid; i < 32768; i += 256) C[i] = (xb[i] - mu) * ninv;
  __syncthreads();

  for (int t = 0; t < Tt; ++t) {
    // ---- content key + beta ----
    if (tid < 32) {
      float acc = bk[tid];
      for (int d = 0; d < DH; ++d) acc += Wk[tid * DH + d] * H[d];
      K[tid] = tanhf(acc);
    } else if (tid == 32) {
      float acc = bbeta[0];
      for (int d = 0; d < DH; ++d) acc += Wbeta[d] * H[d];
      SC[0] = acc > 20.0f ? acc : log1pf(expf(acc));     // softplus
    }
    __syncthreads();
    if (tid == 0) {
      float kn = 0.0f;
      for (int m = 0; m < 32; ++m) kn += K[m] * K[m];
      SC[1] = sqrtf(kn) + 1e-8f;
    }
    __syncthreads();
    // ---- cosine sim + softmax over N ----
    const float beta = SC[0], knorm = SC[1];
    float logit[4];
    float lmax = -3.4e38f;
#pragma unroll
    for (int q = 0; q < 4; ++q) {
      const int n = tid * 4 + q;
      const float* cn = &C[n * 32];
      float dot = 0.0f, nn = 0.0f;
      for (int m = 0; m < 32; ++m) { dot += cn[m] * K[m]; nn += cn[m] * cn[m]; }
      logit[q] = beta * (dot / ((sqrtf(nn) + 1e-8f) * knorm));
      lmax = fmaxf(lmax, logit[q]);
    }
    const float gmax = block_reduce_max(lmax, RED, tid);
    float lsum = 0.0f;
#pragma unroll
    for (int q = 0; q < 4; ++q) { logit[q] = expf(logit[q] - gmax); lsum += logit[q]; }
    const float gsum = block_reduce_sum(lsum, RED, tid);
    const float invs = 1.0f / gsum;
#pragma unroll
    for (int q = 0; q < 4; ++q) W[tid * 4 + q] = logit[q] * invs;
    __syncthreads();
    // ---- read vector r = w . c ----
    {
      const int m = tid & 31, g = tid >> 5;
      float acc = 0.0f;
      for (int n = g * 128; n < (g + 1) * 128; ++n) acc += W[n] * C[n * 32 + m];
      PART[g * 32 + m] = acc;
    }
    __syncthreads();
    if (tid < 32) {
      float acc = 0.0f;
      for (int g = 0; g < 8; ++g) acc += PART[g * 32 + tid];
      R[tid] = acc;
    }
    __syncthreads();
    // ---- controller h_new = tanh(Wh [h; r; xmean] + bh) ----
    const float* xm = xmean + (size_t)(b * Tt + t) * 32;
    float hv;
    {
      float acc = bh[tid];
      const float* wr = Wh + (size_t)tid * 320;
      for (int j = 0; j < 256; ++j) acc += wr[j] * H[j];
      for (int j = 0; j < 32; ++j)  acc += wr[256 + j] * R[j];
      for (int j = 0; j < 32; ++j)  acc += wr[288 + j] * xm[j];
      hv = tanhf(acc);
    }
    __syncthreads();
    H[tid] = hv;
    __syncthreads();
    // ---- gates ----
    if (tid < 32) {
      float ae = 0.0f, aa = 0.0f;
      for (int d = 0; d < DH; ++d) { ae += We[tid * DH + d] * H[d]; aa += Wa[tid * DH + d] * H[d]; }
      E[tid]  = 1.0f / (1.0f + expf(-ae));
      AA[tid] = tanhf(aa);
    } else if (tid == 32) {
      float ag = 0.0f;
      for (int d = 0; d < DH; ++d) ag += Wg[d] * H[d];
      SC[2] = 1.0f / (1.0f + expf(-ag));
    }
    __syncthreads();
    // ---- erase/add write + input gate ----
    const float g = SC[2];
    const float* xt = xb + (size_t)t * 32768;
    float* ct = cb + (size_t)t * 32768;
#pragma unroll
    for (int q = 0; q < 4; ++q) {
      const int n = tid * 4 + q;
      const float wn = W[n];
      for (int m = 0; m < 32; ++m) {
        float cv = C[n * 32 + m];
        float cw = cv * (1.0f - wn * E[m]) + wn * AA[m];
        float cnew = (1.0f - g) * cw + g * xt[n * 32 + m];
        C[n * 32 + m] = cnew;
        ct[n * 32 + m] = cnew;
      }
    }
    __syncthreads();
  }
}

// ---------------- 32-pt radix-2 FFT passes on LDS (32 lines at once) ----------------
__device__ void fft32_pass(float* Re, float* Im, int tid, int lineStride, int tStride) {
  float tr[4], ti[4];
#pragma unroll
  for (int q = 0; q < 4; ++q) {
    int i = tid * 4 + q;
    int l = i >> 5, t = i & 31;
    int rt = (int)(__brev((unsigned)t) >> 27);   // 5-bit bit-reverse
    tr[q] = Re[l * lineStride + rt * tStride];
    ti[q] = Im[l * lineStride + rt * tStride];
  }
  __syncthreads();
#pragma unroll
  for (int q = 0; q < 4; ++q) {
    int i = tid * 4 + q;
    int l = i >> 5, t = i & 31;
    Re[l * lineStride + t * tStride] = tr[q];
    Im[l * lineStride + t * tStride] = ti[q];
  }
  __syncthreads();
  for (int st = 0; st < 5; ++st) {
    const int half = 1 << st;
#pragma unroll
    for (int u = 0; u < 2; ++u) {
      int bidx = tid * 2 + u;              // 512 butterflies/stage
      int l = bidx >> 4, bb = bidx & 15;
      int k = bb & (half - 1);
      int j = ((bb >> st) << (st + 1)) + k;
      int i0 = l * lineStride + j * tStride;
      int i1 = i0 + half * tStride;
      float ang = -3.14159265358979f * (float)k / (float)half;
      float sn, cs; __sincosf(ang, &sn, &cs);
      float xr = Re[i1], xi = Im[i1];
      float wr = cs * xr - sn * xi;
      float wi = cs * xi + sn * xr;
      float ur = Re[i0], ui = Im[i0];
      Re[i0] = ur + wr; Im[i0] = ui + wi;
      Re[i1] = ur - wr; Im[i1] = ui - wi;
    }
    __syncthreads();
  }
}
__device__ __forceinline__ void fft2_inplace(float* Re, float* Im, int tid) {
  fft32_pass(Re, Im, tid, LD, 1);    // rows
  fft32_pass(Re, Im, tid, 1, LD);    // cols
}

// ---------------- DCF correlation: per-sample fused FFT / spectra / inverse ----------------
__global__ void dcf_kernel(const float* __restrict__ c_seq, const float* __restrict__ zf2,
                           const float* __restrict__ cosw, const float* __restrict__ yf,
                           float* __restrict__ out) {
  __shared__ float CW[1024];
  __shared__ float AR[32 * LD], AI[32 * LD];
  __shared__ float BR[32 * LD], BI[32 * LD];
  __shared__ float KZ[32 * WF], KXR[32 * WF], KXI[32 * WF];
  const int s = blockIdx.x, tid = threadIdx.x;
  for (int i = tid; i < 1024; i += 256) CW[i] = cosw[i];
  for (int i = tid; i < 32 * WF; i += 256) { KZ[i] = 0.0f; KXR[i] = 0.0f; KXI[i] = 0.0f; }
  __syncthreads();
  const float* cbase = c_seq + (size_t)s * NSLOT * 32;
  const float* zbase = zf2 + (size_t)s * NSLOT * 32;
  for (int ch = 0; ch < 32; ++ch) {
    // async gather of this channel's 32x32 planes straight into LDS (ASYNCcnt path)
#pragma unroll
    for (int q = 0; q < 4; ++q) {
      const int i = tid * 4 + q;
      const int y = i >> 5, x = i & 31;
      unsigned la = lds_addr32(&AR[y * LD + x]);
      unsigned long long ga = (unsigned long long)(size_t)(cbase + (size_t)i * 32 + ch);
      asm volatile("global_load_async_to_lds_b32 %0, %1, off" :: "v"(la), "v"(ga) : "memory");
      unsigned lb = lds_addr32(&BR[y * LD + x]);
      unsigned long long gb = (unsigned long long)(size_t)(zbase + (size_t)i * 32 + ch);
      asm volatile("global_load_async_to_lds_b32 %0, %1, off" :: "v"(lb), "v"(gb) : "memory");
    }
    asm volatile("s_wait_asynccnt 0" ::: "memory");
    __syncthreads();
    // apply cosine window in place, zero imaginary planes
    for (int i = tid; i < 1024; i += 256) {
      int y = i >> 5, x = i & 31;
      float w = CW[i];
      AR[y * LD + x] *= w; AI[y * LD + x] = 0.0f;
      BR[y * LD + x] *= w; BI[y * LD + x] = 0.0f;
    }
    __syncthreads();
    fft2_inplace(AR, AI, tid);   // cfft
    fft2_inplace(BR, BI, tid);   // zfft
    for (int i = tid; i < 32 * WF; i += 256) {
      int ky = i / WF, kx = i % WF;
      float ar = AR[ky * LD + kx], ai = AI[ky * LD + kx];
      float br = BR[ky * LD + kx], bi = BI[ky * LD + kx];
      KZ[i]  += ar * ar + ai * ai;          // |cfft|^2
      KXR[i] += br * ar + bi * ai;          // z * conj(c)
      KXI[i] += bi * ar - br * ai;
    }
    __syncthreads();
  }
  // S = kxzf * yf / (kzzf + lambda)
  for (int i = tid; i < 32 * WF; i += 256) {
    int ky = i / WF, kx = i % WF;
    float denom = 1.0f / (KZ[i] + 1e-4f);
    float yr = yf[i * 2 + 0], yi = yf[i * 2 + 1];
    float ar = KXR[i], ai = KXI[i];
    AR[ky * LD + kx] = (ar * yr - ai * yi) * denom;
    AI[ky * LD + kx] = (ar * yi + ai * yr) * denom;
  }
  __syncthreads();
  // Hermitian extension kx = 17..31
  for (int i = tid; i < 32 * 15; i += 256) {
    int ky = i / 15, kx = 17 + (i % 15);
    int sy = (32 - ky) & 31, sx = 32 - kx;
    AR[ky * LD + kx] =  AR[sy * LD + sx];
    AI[ky * LD + kx] = -AI[sy * LD + sx];
  }
  __syncthreads();
  // irfft2 via conj trick: real(ifft2(S)) = real(fft2(conj(S)))/1024
  for (int i = tid; i < 1024; i += 256) {
    int y = i >> 5, x = i & 31;
    AI[y * LD + x] = -AI[y * LD + x];
  }
  __syncthreads();
  fft2_inplace(AR, AI, tid);
  float* o = out + (size_t)s * 1024;
  for (int i = tid; i < 1024; i += 256) {
    int y = i >> 5, x = i & 31;
    o[i] = AR[y * LD + x] * (1.0f / 1024.0f);
  }
}

// ---------------- host launcher ----------------
extern "C" void kernel_launch(void* const* d_in, const int* in_sizes, int n_in,
                              void* d_out, int out_size, void* d_ws, size_t ws_size,
                              hipStream_t stream) {
  const float* x_i     = (const float*)d_in[0];
  const float* z_i     = (const float*)d_in[1];
  const float* conv1_w = (const float*)d_in[2];
  const float* conv1_b = (const float*)d_in[3];
  const float* conv2_w = (const float*)d_in[4];
  const float* conv2_b = (const float*)d_in[5];
  const float* Wk      = (const float*)d_in[6];
  const float* bk      = (const float*)d_in[7];
  const float* Wbeta   = (const float*)d_in[8];
  const float* bbeta   = (const float*)d_in[9];
  const float* Wh      = (const float*)d_in[10];
  const float* bh      = (const float*)d_in[11];
  const float* We      = (const float*)d_in[12];
  const float* Wa      = (const float*)d_in[13];
  const float* Wg      = (const float*)d_in[14];
  const float* cosw    = (const float*)d_in[15];
  const float* yf      = (const float*)d_in[16];
  float* out           = (float*)d_out;

  // workspace layout (bytes)
  const size_t y1_elems  = (size_t)BT * 64 * 64 * 32;      // f16
  const size_t f2_elems  = (size_t)BT * NSLOT * 32;        // f32
  char* ws = (char*)d_ws;
  _Float16* y1x = (_Float16*)ws;                 ws += y1_elems * sizeof(_Float16);
  _Float16* y1z = (_Float16*)ws;                 ws += y1_elems * sizeof(_Float16);
  float* xf2    = (float*)ws;                    ws += f2_elems * sizeof(float);
  float* zf2    = (float*)ws;                    ws += f2_elems * sizeof(float);
  float* c_seq  = (float*)ws;                    ws += f2_elems * sizeof(float);
  float* xmean  = (float*)ws;                    ws += (size_t)Bb * Tt * 32 * sizeof(float);
  if ((size_t)(ws - (char*)d_ws) > ws_size) return;

  conv1_kernel<<<dim3(256, BT, 2), 32, 0, stream>>>(x_i, z_i, conv1_w, conv1_b, y1x, y1z);
  conv2_kernel<<<dim3(64, BT, 2), 32, 0, stream>>>(y1x, y1z, conv2_w, conv2_b, xf2, zf2);
  xmean_kernel<<<dim3(BT), 256, 0, stream>>>(xf2, xmean);
  scan_kernel<<<dim3(Bb), 256, SCAN_SMEM_FLOATS * sizeof(float), stream>>>(
      xf2, Wk, bk, Wbeta, bbeta, Wh, bh, We, Wa, Wg, xmean, c_seq);
  dcf_kernel<<<dim3(BT), 256, 0, stream>>>(c_seq, zf2, cosw, yf, out);
}
